// HeteroNet_21543555956876
// MI455X (gfx1250) — compile-verified, matrix operand order checked by reading, not run on hardware
//
#include <hip/hip_runtime.h>
#include <hip/hip_bf16.h>

// ---------------- problem constants (match reference) ----------------
#define NC   30000
#define NT   6000
#define E_CC 300000
#define E_TC 300000
#define E_TT 60000
#define E_CT 60000
#define HID  256
#define OUT  64
#define FEAT 2048

typedef __attribute__((ext_vector_type(16))) _Float16 v16h;
typedef __attribute__((ext_vector_type(8)))  _Float16 v8h;
typedef __attribute__((ext_vector_type(8)))  float    v8f;

// ---------------------------------------------------------------------
// CDNA5 async global->LDS copy helpers (gfx1250, ASYNCcnt-tracked).
// LDS operand is the low-32-bit DS byte offset (generic shared pointers
// on AMDGPU carry the DS offset in their low 32 bits).
// ---------------------------------------------------------------------
__device__ __forceinline__ unsigned lds_off32(const void* p) {
    return (unsigned)(unsigned long long)(uintptr_t)p;
}
__device__ __forceinline__ void async_load_to_lds_b128(unsigned lds_off, const void* gptr) {
    asm volatile("global_load_async_to_lds_b128 %0, %1, off"
                 :: "v"(lds_off), "v"((unsigned long long)(uintptr_t)gptr)
                 : "memory");
}
__device__ __forceinline__ void wait_asynccnt0() {
    asm volatile("s_wait_asynccnt 0x0" ::: "memory");
}

// ---------------------------------------------------------------------
// Tiled GEMM:  C[M,N] = act( A[M,K] @ W[K,N] + bias + beta*C )
//   A: fp32 row-major (converted to f16 while staging to LDS)
//   Wt: f16, PRE-TRANSPOSED to [N,K] row-major  -> B tile staged to LDS
//       column-major (lBt[col][k]) via async b128 copies, so every WMMA
//       fragment is two contiguous ds_load_b128s.
// Block: 256 threads = 8 waves, block tile 128(M) x 128(N).
// Wave tile 32x64: 2 A-frags x 4 B-frags = 8 x v_wmma_f32_16x16x32_f16
// per K-step. Requires K % 32 == 0 (all call sites: 2048/256/512).
// act: 0 = none, 1 = exact GELU.
// ---------------------------------------------------------------------
__global__ __launch_bounds__(256)
void wmma_gemm_kernel(const float* __restrict__ A, const _Float16* __restrict__ Wt,
                      const float* __restrict__ bias, float* __restrict__ C,
                      int M, int N, int K, int beta, int act)
{
    __shared__ __align__(16) _Float16 lA[128 * 32];    // 8 KB, row-major [row][k]
    __shared__ __align__(16) _Float16 lBt[128 * 32];   // 8 KB, col-major [col][k]

    const int tid   = threadIdx.x;
    const int lane  = tid & 31;
    const int wave  = tid >> 5;
    const int half  = lane >> 4;          // 0: lanes 0-15, 1: lanes 16-31
    const int l15   = lane & 15;
    const int waveM = wave & 3;           // 4 wave-rows of 32
    const int waveN = wave >> 2;          // 2 wave-cols of 64
    const int m0    = blockIdx.x * 128;
    const int n0    = blockIdx.y * 128;

    v8f acc[2][4];
    #pragma unroll
    for (int s = 0; s < 2; ++s)
        #pragma unroll
        for (int j = 0; j < 4; ++j)
            acc[s][j] = (v8f){};

    for (int k0 = 0; k0 < K; k0 += 32) {
        // ---- stage A tile (128 rows x 32 k), fp32 -> f16, 16 elems/thread ----
        {
            const int r   = tid >> 1;
            const int kb  = (tid & 1) * 16;
            const int row = m0 + r;
            v8h t0, t1;
            if (row < M) {
                const float* ap = A + (size_t)row * K + k0 + kb;
                #pragma unroll
                for (int j = 0; j < 8; ++j) {
                    t0[j] = (_Float16)ap[j];
                    t1[j] = (_Float16)ap[8 + j];
                }
            } else {
                t0 = (v8h){}; t1 = (v8h){};
            }
            *(v8h*)&lA[r * 32 + kb]     = t0;
            *(v8h*)&lA[r * 32 + kb + 8] = t1;
        }
        // ---- stage B tile (128 cols x 32 k) via async global->LDS b128 ----
        // 512 16-byte chunks; each thread copies 2.
        {
            #pragma unroll
            for (int c = 0; c < 2; ++c) {
                const int chunk = tid + c * 256;
                const int col   = chunk >> 2;        // 0..127
                const int ke    = (chunk & 3) * 8;   // f16 element offset in k
                _Float16* dstp  = &lBt[col * 32 + ke];
                if (n0 + col < N) {
                    const _Float16* srcp = Wt + (size_t)(n0 + col) * K + k0 + ke;
                    async_load_to_lds_b128(lds_off32(dstp), srcp);
                } else {
                    *(v8h*)dstp = (v8h){};
                }
            }
        }
        wait_asynccnt0();       // my async copies landed in LDS
        __syncthreads();        // everyone's staging visible

        // ---- fragments: contiguous b128 pairs per ISA 16-bit layout ----
        // element i -> k = (i&7) + half*8 + (i>=8)*16
        union V16 { v16h v; v8h h[2]; };
        v16h af[2], bf[4];
        #pragma unroll
        for (int s = 0; s < 2; ++s) {
            const int arow = waveM * 32 + s * 16 + l15;
            const _Float16* ab = &lA[arow * 32 + (half << 3)];
            V16 u;
            u.h[0] = *(const v8h*)(ab);
            u.h[1] = *(const v8h*)(ab + 16);
            af[s] = u.v;
        }
        #pragma unroll
        for (int j = 0; j < 4; ++j) {
            const int bcol = waveN * 64 + j * 16 + l15;
            const _Float16* bb = &lBt[bcol * 32 + (half << 3)];
            V16 u;
            u.h[0] = *(const v8h*)(bb);
            u.h[1] = *(const v8h*)(bb + 16);
            bf[j] = u.v;
        }

        #pragma unroll
        for (int s = 0; s < 2; ++s)
            #pragma unroll
            for (int j = 0; j < 4; ++j)
                acc[s][j] = __builtin_amdgcn_wmma_f32_16x16x32_f16(
                    false, af[s], false, bf[j], (short)0, acc[s][j], false, false);

        __syncthreads();        // protect LDS before next staging
    }

    // ---- epilogue: per 16x16 tile, VGPR r holds M = r + half*8, N = l15 ----
    #pragma unroll
    for (int j = 0; j < 4; ++j) {
        const int col = n0 + waveN * 64 + j * 16 + l15;
        if (col >= N) continue;
        const float bv = bias ? bias[col] : 0.0f;
        #pragma unroll
        for (int s = 0; s < 2; ++s) {
            const int rbase = m0 + waveM * 32 + s * 16 + (half ? 8 : 0);
            #pragma unroll
            for (int r = 0; r < 8; ++r) {
                const int row = rbase + r;
                if (row < M) {
                    float v = acc[s][j][r] + bv;
                    float* cp = C + (size_t)row * N + col;
                    if (beta) v += *cp;
                    if (act == 1)
                        v = 0.5f * v * (1.0f + erff(v * 0.70710678118654752f));
                    *cp = v;
                }
            }
        }
    }
}

// ---------------------------------------------------------------------
// Convert fp32 [B][K][N] -> f16 transposed [B][N][K] (weights, once/call)
// ---------------------------------------------------------------------
__global__ void transpose_f16_kernel(const float* __restrict__ in,
                                     _Float16* __restrict__ out,
                                     int Bn, int K, int N)
{
    int idx = blockIdx.x * blockDim.x + threadIdx.x;
    const int total = Bn * K * N;
    if (idx < total) {
        const int b   = idx / (K * N);
        const int rem = idx - b * K * N;
        const int n   = rem / K;
        const int k   = rem - n * K;
        out[idx] = (_Float16)in[(size_t)b * K * N + (size_t)k * N + n];
    }
}

// ---------------------------------------------------------------------
// Edge in-degree counts (layer-independent; computed once per call)
// ---------------------------------------------------------------------
__global__ void edge_count_kernel(const int* __restrict__ dst,
                                  float* __restrict__ cnt, int E)
{
    int e = blockIdx.x * blockDim.x + threadIdx.x;
    if (e < E) atomicAdd(&cnt[dst[e]], 1.0f);
}

// ---------------------------------------------------------------------
// Edge scatter: acc[dst] += ew * h_src[src]. One wave per edge.
// ---------------------------------------------------------------------
__global__ __launch_bounds__(256)
void edge_scatter_kernel(const float* __restrict__ hsrc,
                         const int* __restrict__ src,
                         const int* __restrict__ dst,
                         const float* __restrict__ ew,
                         float* __restrict__ acc, int E, int F)
{
    const int e = blockIdx.x * 8 + (threadIdx.x >> 5);
    if (e >= E) return;
    const int lane = threadIdx.x & 31;
    const int s = src[e];
    const int d = dst[e];
    const float w = ew[e];
    const float* hp = hsrc + (size_t)s * F;
    float* ap = acc + (size_t)d * F;
    for (int f = lane; f < F; f += 32)
        atomicAdd(&ap[f], w * hp[f]);
}

// ---------------------------------------------------------------------
// In-place segment mean: acc[i,f] /= max(cnt[i], 1)
// ---------------------------------------------------------------------
__global__ void seg_mean_kernel(float* __restrict__ acc,
                                const float* __restrict__ cnt, int N, int F)
{
    int idx = blockIdx.x * blockDim.x + threadIdx.x;
    if (idx < N * F) {
        const int row = idx / F;
        float c = cnt[row];
        if (c < 1.0f) c = 1.0f;
        acc[idx] /= c;
    }
}

// ---------------------------------------------------------------------
// Fused: out = gelu( 0.5*GN(lrelu(o1);g1,b1) + 0.5*GN(lrelu(o2);g2,b2) )
// GroupNorm C=256, 4 groups (64 ch) = 8 lanes x 8 contiguous features.
// ---------------------------------------------------------------------
__global__ __launch_bounds__(256)
void agg_gn_gelu_kernel(const float* __restrict__ o1, const float* __restrict__ o2,
                        const float* __restrict__ g1, const float* __restrict__ b1,
                        const float* __restrict__ g2, const float* __restrict__ b2,
                        float* __restrict__ out, int Nrows)
{
    const int row = blockIdx.x * 8 + (threadIdx.x >> 5);
    if (row >= Nrows) return;
    const int lane = threadIdx.x & 31;
    const int fb = lane * 8;

    float x1[8], x2[8];
    float s1 = 0.f, q1 = 0.f, s2 = 0.f, q2 = 0.f;
    const float* p1 = o1 + (size_t)row * HID + fb;
    const float* p2 = o2 + (size_t)row * HID + fb;
    #pragma unroll
    for (int j = 0; j < 8; ++j) {
        float a = p1[j]; a = (a > 0.f) ? a : 0.01f * a;  // leaky_relu
        float c = p2[j]; c = (c > 0.f) ? c : 0.01f * c;
        x1[j] = a; s1 += a; q1 += a * a;
        x2[j] = c; s2 += c; q2 += c * c;
    }
    #pragma unroll
    for (int m = 1; m < 8; m <<= 1) {
        s1 += __shfl_xor(s1, m, 8);
        q1 += __shfl_xor(q1, m, 8);
        s2 += __shfl_xor(s2, m, 8);
        q2 += __shfl_xor(q2, m, 8);
    }
    const float inv = 1.0f / 64.0f;
    const float mu1 = s1 * inv, var1 = q1 * inv - mu1 * mu1;
    const float mu2 = s2 * inv, var2 = q2 * inv - mu2 * mu2;
    const float r1 = rsqrtf(var1 + 1e-5f);
    const float r2 = rsqrtf(var2 + 1e-5f);

    float* po = out + (size_t)row * HID + fb;
    #pragma unroll
    for (int j = 0; j < 8; ++j) {
        const int f = fb + j;
        const float n1 = (x1[j] - mu1) * r1 * g1[f] + b1[f];
        const float n2 = (x2[j] - mu2) * r2 * g2[f] + b2[f];
        float v = 0.5f * n1 + 0.5f * n2;
        v = 0.5f * v * (1.0f + erff(v * 0.70710678118654752f));
        po[j] = v;
    }
}

// ---------------------------------------------------------------------
// Concat along features: out[r, 0:F] = a[r], out[r, F:2F] = b[r]
// ---------------------------------------------------------------------
__global__ void concat2_kernel(const float* __restrict__ a,
                               const float* __restrict__ b,
                               float* __restrict__ out, int N, int F)
{
    int idx = blockIdx.x * blockDim.x + threadIdx.x;
    if (idx < N * F) {
        const int r = idx / F;
        const int f = idx - r * F;
        out[(size_t)r * 2 * F + f]     = a[idx];
        out[(size_t)r * 2 * F + F + f] = b[idx];
    }
}

// ---------------------------------------------------------------------
// log_softmax over 64 columns. One wave per row, 2 values per lane.
// ---------------------------------------------------------------------
__global__ __launch_bounds__(256)
void log_softmax64_kernel(const float* __restrict__ in,
                          float* __restrict__ out, int N)
{
    const int row = blockIdx.x * 8 + (threadIdx.x >> 5);
    if (row >= N) return;
    const int lane = threadIdx.x & 31;
    const float a = in[(size_t)row * OUT + lane];
    const float b = in[(size_t)row * OUT + 32 + lane];
    float mx = fmaxf(a, b);
    #pragma unroll
    for (int m = 16; m > 0; m >>= 1) mx = fmaxf(mx, __shfl_xor(mx, m, 32));
    float es = expf(a - mx) + expf(b - mx);
    #pragma unroll
    for (int m = 16; m > 0; m >>= 1) es += __shfl_xor(es, m, 32);
    const float l = mx + logf(es);
    out[(size_t)row * OUT + lane]      = a - l;
    out[(size_t)row * OUT + 32 + lane] = b - l;
}

// ---------------------------------------------------------------------
// Host orchestration
// ---------------------------------------------------------------------
static inline void launch_gemm(const float* A, const _Float16* Wt, const float* bias,
                               float* C, int M, int N, int K, int beta, int act,
                               hipStream_t s)
{
    dim3 grid((M + 127) / 128, (N + 127) / 128);
    wmma_gemm_kernel<<<grid, 256, 0, s>>>(A, Wt, bias, C, M, N, K, beta, act);
}

extern "C" void kernel_launch(void* const* d_in, const int* in_sizes, int n_in,
                              void* d_out, int out_size, void* d_ws, size_t ws_size,
                              hipStream_t stream)
{
    (void)in_sizes; (void)n_in; (void)out_size; (void)ws_size;

    // ---- inputs (setup_inputs dict order) ----
    const float* cell_feat   = (const float*)d_in[0];
    const float* tissue_feat = (const float*)d_in[1];
    const float* lin1_w = (const float*)d_in[2];
    const float* lin1_b = (const float*)d_in[3];
    const float* lin2_w = (const float*)d_in[4];
    const float* lin2_b = (const float*)d_in[5];
    const float* ew_c2c = (const float*)d_in[6];
    const float* ew_t2c = (const float*)d_in[7];
    const float* ew_t2t = (const float*)d_in[8];
    const float* ew_c2t = (const float*)d_in[9];
    const float* wself0  = (const float*)d_in[10];
    const float* wneigh0 = (const float*)d_in[11];
    const float* bias0   = (const float*)d_in[12];
    const float* wself1  = (const float*)d_in[13];
    const float* wneigh1 = (const float*)d_in[14];
    const float* bias1   = (const float*)d_in[15];
    const float* gn_gamma = (const float*)d_in[16];
    const float* gn_beta  = (const float*)d_in[17];
    const float* ro_w0 = (const float*)d_in[18];
    const float* ro_b0 = (const float*)d_in[19];
    const float* ro_w1 = (const float*)d_in[20];
    const float* ro_b1 = (const float*)d_in[21];
    const int* src_c2c = (const int*)d_in[22];
    const int* dst_c2c = (const int*)d_in[23];
    const int* src_t2c = (const int*)d_in[24];
    const int* dst_t2c = (const int*)d_in[25];
    const int* src_t2t = (const int*)d_in[26];
    const int* dst_t2t = (const int*)d_in[27];
    const int* src_c2t = (const int*)d_in[28];
    const int* dst_c2t = (const int*)d_in[29];

    float* outp = (float*)d_out;

    // ---- workspace layout (floats) ----
    const size_t NCxH  = (size_t)NC * HID;
    const size_t NTxH  = (size_t)NT * HID;
    const size_t NCx2H = (size_t)NC * 2 * HID;
    const size_t NTx2H = (size_t)NT * 2 * HID;

    float* p = (float*)d_ws;
    float* hc0    = p; p += NCxH;     // initial cell projection
    float* ht0    = p; p += NTxH;     // initial tissue projection
    float* hcA    = p; p += NCxH;     // layer-0 cell output (hist[0])
    float* htA    = p; p += NTxH;     // layer-0 tissue output
    float* hcB    = p; p += NCxH;     // layer-1 cell output (hist[1])
    float* hcat_c = p; p += NCx2H;    // [hcA | hc0]; reused for hist concat
    float* hcat_t = p; p += NTx2H;    // [htA | ht0]
    float* acc1   = p; p += NCx2H;    // c2c agg; reused for readout hidden
    float* acc2   = p; p += NCx2H;    // t2c agg; reused for logits
    float* acc3   = p; p += NTx2H;    // t2t agg (layer 0 only)
    float* acc4   = p; p += NTx2H;    // c2t agg (layer 0 only)
    float* cnt1   = p; p += NC;
    float* cnt2   = p; p += NC;
    float* cnt3   = p; p += NT;
    float* cnt4   = p; p += NT;
    float* o1     = p; p += NCxH;
    float* o2     = p; p += NCxH;
    float* o3     = p; p += NTxH;
    float* o4     = p; p += NTxH;

    // ---- f16 transposed weights (converted once per call) ----
    _Float16* hptr = (_Float16*)p;
    _Float16* lin1_wt   = hptr; hptr += (size_t)FEAT * HID;
    _Float16* lin2_wt   = hptr; hptr += (size_t)FEAT * HID;
    _Float16* wself0t   = hptr; hptr += (size_t)4 * HID * HID;
    _Float16* wneigh0t  = hptr; hptr += (size_t)4 * HID * HID;
    _Float16* wself1t   = hptr; hptr += (size_t)4 * 2 * HID * HID;
    _Float16* wneigh1t  = hptr; hptr += (size_t)4 * 2 * HID * HID;
    _Float16* ro_w0t    = hptr; hptr += (size_t)2 * HID * 2 * HID;
    _Float16* ro_w1t    = hptr; hptr += (size_t)2 * HID * OUT;

    // ---- zero accumulators / counts ----
    hipMemsetAsync(acc1, 0, NCx2H * sizeof(float), stream);
    hipMemsetAsync(acc2, 0, NCx2H * sizeof(float), stream);
    hipMemsetAsync(acc3, 0, NTx2H * sizeof(float), stream);
    hipMemsetAsync(acc4, 0, NTx2H * sizeof(float), stream);
    hipMemsetAsync(cnt1, 0, NC * sizeof(float), stream);
    hipMemsetAsync(cnt2, 0, NC * sizeof(float), stream);
    hipMemsetAsync(cnt3, 0, NT * sizeof(float), stream);
    hipMemsetAsync(cnt4, 0, NT * sizeof(float), stream);

    // ---- weight conversion: fp32 [K,N] -> f16 [N,K] ----
    {
        auto tlaunch = [&](const float* in, _Float16* out, int B, int K, int N) {
            int total = B * K * N;
            transpose_f16_kernel<<<(total + 255) / 256, 256, 0, stream>>>(in, out, B, K, N);
        };
        tlaunch(lin1_w,  lin1_wt,  1, FEAT, HID);
        tlaunch(lin2_w,  lin2_wt,  1, FEAT, HID);
        tlaunch(wself0,  wself0t,  4, HID, HID);
        tlaunch(wneigh0, wneigh0t, 4, HID, HID);
        tlaunch(wself1,  wself1t,  4, 2 * HID, HID);
        tlaunch(wneigh1, wneigh1t, 4, 2 * HID, HID);
        tlaunch(ro_w0,   ro_w0t,   1, 2 * HID, 2 * HID);
        tlaunch(ro_w1,   ro_w1t,   1, 2 * HID, OUT);
    }

    // ---- in-degree counts (layer-independent) ----
    edge_count_kernel<<<(E_CC + 255) / 256, 256, 0, stream>>>(dst_c2c, cnt1, E_CC);
    edge_count_kernel<<<(E_TC + 255) / 256, 256, 0, stream>>>(dst_t2c, cnt2, E_TC);
    edge_count_kernel<<<(E_TT + 255) / 256, 256, 0, stream>>>(dst_t2t, cnt3, E_TT);
    edge_count_kernel<<<(E_CT + 255) / 256, 256, 0, stream>>>(dst_c2t, cnt4, E_CT);

    // ---- input projections (WMMA GEMM, K=2048) ----
    launch_gemm(cell_feat,   lin1_wt, lin1_b, hc0, NC, HID, FEAT, 0, 0, stream);
    launch_gemm(tissue_feat, lin2_wt, lin2_b, ht0, NT, HID, FEAT, 0, 0, stream);

    // ==================== Layer 0 (F = 256) ====================
    edge_scatter_kernel<<<(E_CC + 7) / 8, 256, 0, stream>>>(hc0, src_c2c, dst_c2c, ew_c2c, acc1, E_CC, HID);
    edge_scatter_kernel<<<(E_TC + 7) / 8, 256, 0, stream>>>(ht0, src_t2c, dst_t2c, ew_t2c, acc2, E_TC, HID);
    edge_scatter_kernel<<<(E_TT + 7) / 8, 256, 0, stream>>>(ht0, src_t2t, dst_t2t, ew_t2t, acc3, E_TT, HID);
    edge_scatter_kernel<<<(E_CT + 7) / 8, 256, 0, stream>>>(hc0, src_c2t, dst_c2t, ew_c2t, acc4, E_CT, HID);
    seg_mean_kernel<<<((int)NCxH + 255) / 256, 256, 0, stream>>>(acc1, cnt1, NC, HID);
    seg_mean_kernel<<<((int)NCxH + 255) / 256, 256, 0, stream>>>(acc2, cnt2, NC, HID);
    seg_mean_kernel<<<((int)NTxH + 255) / 256, 256, 0, stream>>>(acc3, cnt3, NT, HID);
    seg_mean_kernel<<<((int)NTxH + 255) / 256, 256, 0, stream>>>(acc4, cnt4, NT, HID);

    // combine: o = h_dst @ ws + mean_neigh @ wn + b   (two WMMA passes)
    launch_gemm(hc0,  wself0t  + 0 * HID * HID, bias0 + 0 * HID, o1, NC, HID, HID, 0, 0, stream);
    launch_gemm(acc1, wneigh0t + 0 * HID * HID, nullptr,         o1, NC, HID, HID, 1, 0, stream);
    launch_gemm(hc0,  wself0t  + 1 * HID * HID, bias0 + 1 * HID, o2, NC, HID, HID, 0, 0, stream);
    launch_gemm(acc2, wneigh0t + 1 * HID * HID, nullptr,         o2, NC, HID, HID, 1, 0, stream);
    launch_gemm(ht0,  wself0t  + 2 * HID * HID, bias0 + 2 * HID, o3, NT, HID, HID, 0, 0, stream);
    launch_gemm(acc3, wneigh0t + 2 * HID * HID, nullptr,         o3, NT, HID, HID, 1, 0, stream);
    launch_gemm(ht0,  wself0t  + 3 * HID * HID, bias0 + 3 * HID, o4, NT, HID, HID, 0, 0, stream);
    launch_gemm(acc4, wneigh0t + 3 * HID * HID, nullptr,         o4, NT, HID, HID, 1, 0, stream);

    // attention_agg + gelu (GN params rows 0 and 1)
    agg_gn_gelu_kernel<<<(NC + 7) / 8, 256, 0, stream>>>(
        o1, o2, gn_gamma + 0 * HID, gn_beta + 0 * HID,
        gn_gamma + 1 * HID, gn_beta + 1 * HID, hcA, NC);
    agg_gn_gelu_kernel<<<(NT + 7) / 8, 256, 0, stream>>>(
        o3, o4, gn_gamma + 0 * HID, gn_beta + 0 * HID,
        gn_gamma + 1 * HID, gn_beta + 1 * HID, htA, NT);

    // res_cat with initial residual: [h, h0]
    concat2_kernel<<<((int)NCxH + 255) / 256, 256, 0, stream>>>(hcA, hc0, hcat_c, NC, HID);
    concat2_kernel<<<((int)NTxH + 255) / 256, 256, 0, stream>>>(htA, ht0, hcat_t, NT, HID);

    // ============ Layer 1 (F = 512; only cell-dst convs are live) ============
    hipMemsetAsync(acc1, 0, NCx2H * sizeof(float), stream);
    hipMemsetAsync(acc2, 0, NCx2H * sizeof(float), stream);
    edge_scatter_kernel<<<(E_CC + 7) / 8, 256, 0, stream>>>(hcat_c, src_c2c, dst_c2c, ew_c2c, acc1, E_CC, 2 * HID);
    edge_scatter_kernel<<<(E_TC + 7) / 8, 256, 0, stream>>>(hcat_t, src_t2c, dst_t2c, ew_t2c, acc2, E_TC, 2 * HID);
    seg_mean_kernel<<<((int)NCx2H + 255) / 256, 256, 0, stream>>>(acc1, cnt1, NC, 2 * HID);
    seg_mean_kernel<<<((int)NCx2H + 255) / 256, 256, 0, stream>>>(acc2, cnt2, NC, 2 * HID);

    launch_gemm(hcat_c, wself1t  + 0 * 2 * HID * HID, bias1 + 0 * HID, o1, NC, HID, 2 * HID, 0, 0, stream);
    launch_gemm(acc1,   wneigh1t + 0 * 2 * HID * HID, nullptr,         o1, NC, HID, 2 * HID, 1, 0, stream);
    launch_gemm(hcat_c, wself1t  + 1 * 2 * HID * HID, bias1 + 1 * HID, o2, NC, HID, 2 * HID, 0, 0, stream);
    launch_gemm(acc2,   wneigh1t + 1 * 2 * HID * HID, nullptr,         o2, NC, HID, 2 * HID, 1, 0, stream);

    // GN params rows 2 and 3
    agg_gn_gelu_kernel<<<(NC + 7) / 8, 256, 0, stream>>>(
        o1, o2, gn_gamma + 2 * HID, gn_beta + 2 * HID,
        gn_gamma + 3 * HID, gn_beta + 3 * HID, hcB, NC);

    // ==================== Readout ====================
    concat2_kernel<<<((int)NCxH + 255) / 256, 256, 0, stream>>>(hcA, hcB, hcat_c, NC, HID);

    // hidden = gelu(hist @ ro_w0 + ro_b0)   (reuse acc1 as [NC, 512])
    launch_gemm(hcat_c, ro_w0t, ro_b0, acc1, NC, 2 * HID, 2 * HID, 0, 1, stream);
    // logits = hidden @ ro_w1 + ro_b1       (reuse acc2 as [NC, 64])
    launch_gemm(acc1, ro_w1t, ro_b1, acc2, NC, OUT, 2 * HID, 0, 0, stream);

    // log_softmax -> d_out
    log_softmax64_kernel<<<(NC + 7) / 8, 256, 0, stream>>>(acc2, outp, NC);
}
